// Model_19653770346966
// MI455X (gfx1250) — matrix operand order repaired and checked by reading, not compiled, not run on hardware
//
#include <hip/hip_runtime.h>
#include <hip/hip_bf16.h>

// ---------------- problem constants ----------------
static constexpr int N_TOK = 8192;
static constexpr int DM    = 512;     // d_model
static constexpr int NE    = 8;       // experts
// top_k == 2 hardcoded (reference constant)

// ---------------- workspace layout (bytes) ----------------
static constexpr size_t OFF_COMBINED = 0;                       // float [N_TOK*DM]  = 16 MB
static constexpr size_t OFF_XHI      = 16u * 1024 * 1024;       // bf16  [N_TOK*DM]  =  8 MB
static constexpr size_t OFF_XLO      = 24u * 1024 * 1024;       // bf16  [N_TOK*DM]  =  8 MB
static constexpr size_t OFF_WHT      = 32u * 1024 * 1024;       // bf16  [NE*DM*DM] T =  4 MB
static constexpr size_t OFF_WLT      = 36u * 1024 * 1024;       // bf16  [NE*DM*DM] T =  4 MB
static constexpr size_t OFF_CNT      = 40u * 1024 * 1024;       // int   [NE]
static constexpr size_t OFF_IMP      = OFF_CNT + 256;           // float [NE]
static constexpr size_t OFF_LDC      = OFF_CNT + 512;           // float [NE]
static constexpr size_t OFF_BTOK     = OFF_CNT + 4096;          // int   [NE*N_TOK] = 256 KB
static constexpr size_t OFF_BGATE    = OFF_BTOK + (size_t)NE * N_TOK * 4; // float [NE*N_TOK]

typedef __attribute__((ext_vector_type(16))) __bf16 v16bf;
typedef __attribute__((ext_vector_type(8)))  __bf16 v8bf;
typedef __attribute__((ext_vector_type(8)))  float  v8f;

// ---------------- fp32 -> bf16 hi/lo split (x ~= hi + lo) ----------------
__device__ __forceinline__ void bf16_split(float v, __bf16& h, __bf16& l) {
    h = (__bf16)v;
    l = (__bf16)(v - (float)h);
}

__device__ __forceinline__ v16bf load_frag16(const __bf16* p0, const __bf16* p1) {
    union { v16bf v; v8bf h[2]; } u;
    u.h[0] = *reinterpret_cast<const v8bf*>(p0);
    u.h[1] = *reinterpret_cast<const v8bf*>(p1);
    return u.v;
}

// ---------------- CDNA5 async copy: global -> LDS, 16B per lane, ASYNCcnt ----------------
__device__ __forceinline__ uint32_t lds_off_u32(const void* p) {
    // generic -> addrspace(3) cast yields the raw LDS byte offset
    return (uint32_t)(uintptr_t)(__attribute__((address_space(3))) const void*)p;
}

__device__ __forceinline__ void async_copy_b128(const void* gsrc, void* ldst) {
    uint32_t loff = lds_off_u32(ldst);
    asm volatile("global_load_async_to_lds_b128 %0, %1, off"
                 :: "v"(loff), "v"(gsrc) : "memory");
}

__device__ __forceinline__ void wait_asynccnt0() {
    asm volatile("s_wait_asynccnt 0" ::: "memory");
}

// ---------------- kernel: convert x to bf16 hi/lo ----------------
__global__ void cvt_x_kernel(const float* __restrict__ x, __bf16* __restrict__ xh,
                             __bf16* __restrict__ xl, int n) {
    int i = blockIdx.x * blockDim.x + threadIdx.x;
    if (i >= n) return;
    __bf16 h, l;
    bf16_split(x[i], h, l);
    xh[i] = h; xl[i] = l;
}

// ---------------- kernel: convert expert_w to bf16 hi/lo, transposed [e][f][d] ----------------
__global__ void cvt_w_kernel(const float* __restrict__ w, __bf16* __restrict__ wh,
                             __bf16* __restrict__ wl) {
    int i = blockIdx.x * blockDim.x + threadIdx.x;
    if (i >= NE * DM * DM) return;
    int e = i / (DM * DM);
    int r = i - e * (DM * DM);
    int d = r / DM;
    int f = r - d * DM;
    __bf16 h, l;
    bf16_split(w[i], h, l);
    size_t o = ((size_t)e * DM + f) * DM + d;   // transposed: [e][f][d]
    wh[o] = h; wl[o] = l;
}

// ---------------- kernel: gating (logits -> top2 -> softmax -> buckets) ----------------
__global__ void gate_kernel(const float* __restrict__ x, const float* __restrict__ wg,
                            int* __restrict__ counts, float* __restrict__ imp,
                            float* __restrict__ ldc, int* __restrict__ btok,
                            float* __restrict__ bgate) {
    __shared__ float s_wg[DM * NE];          // 16 KB
    for (int i = threadIdx.x; i < DM * NE; i += blockDim.x) s_wg[i] = wg[i];
    __syncthreads();

    int n = blockIdx.x * blockDim.x + threadIdx.x;
    if (n >= N_TOK) return;

    float lg[NE];
#pragma unroll
    for (int e = 0; e < NE; ++e) lg[e] = 0.0f;

    const float4* xr = reinterpret_cast<const float4*>(x + (size_t)n * DM);
    for (int d4 = 0; d4 < DM / 4; ++d4) {
        float4 xv = xr[d4];
        const float xs[4] = {xv.x, xv.y, xv.z, xv.w};
        int d = d4 * 4;
#pragma unroll
        for (int j = 0; j < 4; ++j)
#pragma unroll
            for (int e = 0; e < NE; ++e)
                lg[e] = fmaf(xs[j], s_wg[(d + j) * NE + e], lg[e]);
    }

    // top-2 (distinct indices)
    int i0 = 0; float v0 = lg[0];
#pragma unroll
    for (int e = 1; e < NE; ++e) if (lg[e] > v0) { v0 = lg[e]; i0 = e; }
    int i1 = -1; float v1 = -3.4e38f;
#pragma unroll
    for (int e = 0; e < NE; ++e) if (e != i0 && lg[e] > v1) { v1 = lg[e]; i1 = e; }

    // softmax over {v0, v1}, v0 is the max
    float t  = __expf(v1 - v0);
    float g0 = 1.0f / (1.0f + t);
    float g1 = t * g0;

    int p0 = atomicAdd(&counts[i0], 1);
    btok[i0 * N_TOK + p0]  = n;
    bgate[i0 * N_TOK + p0] = g0;
    int p1 = atomicAdd(&counts[i1], 1);
    btok[i1 * N_TOK + p1]  = n;
    bgate[i1 * N_TOK + p1] = g1;

    atomicAdd(&imp[i0], g0);
    atomicAdd(&imp[i1], g1);
    atomicAdd(&ldc[i0], 1.0f);
    atomicAdd(&ldc[i1], 1.0f);
}

// ---------------- kernel: balance loss (cv^2(importance) + cv^2(load)) ----------------
__global__ void loss_kernel(const float* __restrict__ imp, const float* __restrict__ ldc,
                            float* __restrict__ out) {
    if (threadIdx.x != 0 || blockIdx.x != 0) return;
    float m1 = 0.f, m2 = 0.f;
#pragma unroll
    for (int e = 0; e < NE; ++e) { m1 += imp[e]; m2 += ldc[e]; }
    m1 *= (1.0f / NE); m2 *= (1.0f / NE);
    float s1 = 0.f, s2 = 0.f;
#pragma unroll
    for (int e = 0; e < NE; ++e) {
        float d1 = imp[e] - m1; s1 += d1 * d1;
        float d2 = ldc[e] - m2; s2 += d2 * d2;
    }
    s1 *= (1.0f / NE); s2 *= (1.0f / NE);
    out[0] = s1 / (m1 * m1 + 1e-10f) + s2 / (m2 * m2 + 1e-10f);
}

// ---------------- kernel: bucketed expert GEMM ----------------
// grid: (N_TOK/64 token tiles, DM/128 col groups, NE experts), block 256 = 8 waves.
// Block tile: 64 tokens x 128 cols. A (hi+lo) double-buffer staged in LDS via
// global_load_async_to_lds_b128 (K-chunks of 64, overlapped with compute).
// Each wave: 4 M-subtiles x 16 cols. Per K-step: batch-load 2 B frags (global)
// + 8 A frags (LDS), single wait, then a 12-WMMA chain.
static constexpr int MT     = 64;    // token rows per block
static constexpr int KCHUNK = 64;    // K elements per staged chunk
static constexpr int LDA    = 72;    // padded LDS row stride (bf16 elems) -> conflict-free

__global__ void __launch_bounds__(256)
moe_gemm_kernel(const __bf16* __restrict__ xh, const __bf16* __restrict__ xl,
                const __bf16* __restrict__ whT, const __bf16* __restrict__ wlT,
                const int* __restrict__ counts, const int* __restrict__ btok,
                const float* __restrict__ bgate, float* __restrict__ combined) {
    __shared__ __align__(16) __bf16 s_ah[2][MT][LDA];
    __shared__ __align__(16) __bf16 s_al[2][MT][LDA];
    __shared__ int   s_tok[MT];
    __shared__ float s_gate[MT];

    const int e = blockIdx.z;
    const int cnt = counts[e];
    const int tileBase = blockIdx.x * MT;
    if (tileBase >= cnt) return;

    const int tid      = threadIdx.x;
    const int wave     = tid >> 5;           // 0..7
    const int lane     = tid & 31;
    const int laneHalf = lane >> 4;          // 0 | 1
    const int mrow     = lane & 15;
    const int ncol     = blockIdx.y * 128 + wave * 16 + mrow;

    const __bf16* bcol_h = whT + ((size_t)e * DM + ncol) * DM;
    const __bf16* bcol_l = wlT + ((size_t)e * DM + ncol) * DM;

    // cache gathered token ids + gates for this tile (ragged tail clamped)
    if (tid < MT) {
        int ti = tileBase + tid;
        if (ti >= cnt) ti = cnt - 1;
        s_tok[tid]  = btok[e * N_TOK + ti];
        s_gate[tid] = bgate[e * N_TOK + ti];
    }
    __syncthreads();

    // async-stage one K-chunk of A (hi+lo) into LDS buffer `buf`
    auto issue_chunk = [&](int chunk, int buf) {
        const int kb = chunk * KCHUNK;
        // MT rows * (KCHUNK/8) 16B-pieces * 2 matrices = 1024 async b128 ops / block
        for (int idx = tid; idx < MT * (KCHUNK / 8) * 2; idx += 256) {
            const int mtx = idx >> 9;          // 0 = hi, 1 = lo
            const int row = (idx >> 3) & (MT - 1);
            const int c   = idx & 7;
            const __bf16* g = (mtx ? xl : xh) + (size_t)s_tok[row] * DM + kb + c * 8;
            __bf16* l = mtx ? &s_al[buf][row][c * 8] : &s_ah[buf][row][c * 8];
            async_copy_b128(g, l);
        }
    };

    v8f acc[4];
#pragma unroll
    for (int ms = 0; ms < 4; ++ms) acc[ms] = (v8f){};

    issue_chunk(0, 0);

    for (int ch = 0; ch < DM / KCHUNK; ++ch) {
        wait_asynccnt0();
        __syncthreads();                      // chunk `ch` resident; prev buffer free
        if (ch + 1 < DM / KCHUNK) issue_chunk(ch + 1, (ch + 1) & 1);

        const int buf = ch & 1;
        const int kb  = ch * KCHUNK;
#pragma unroll
        for (int ks = 0; ks < KCHUNK; ks += 32) {
            // ---- batched loads first: 2 B frag pairs (global) + 8 A frags (LDS) ----
            // B 32x16 layout: column = lane&15; lanes 0-15 K in [0,8)+[16,24); lanes 16-31 +8
            const int b0 = kb + ks + laneHalf * 8;
            v16bf bh = load_frag16(bcol_h + b0, bcol_h + b0 + 16);
            v16bf bl = load_frag16(bcol_l + b0, bcol_l + b0 + 16);
            v16bf ah[4], al[4];
#pragma unroll
            for (int ms = 0; ms < 4; ++ms) {
                // A 16x32 layout: lanes 0-15 K in [0,8)+[16,24); lanes 16-31 +8
                const __bf16* ph = &s_ah[buf][ms * 16 + mrow][ks + laneHalf * 8];
                const __bf16* pl = &s_al[buf][ms * 16 + mrow][ks + laneHalf * 8];
                ah[ms] = load_frag16(ph, ph + 16);
                al[ms] = load_frag16(pl, pl + 16);
            }
            // ---- 12-WMMA chain (bf16x3: hi*hi + hi*lo + lo*hi, ~fp32 accuracy) ----
#pragma unroll
            for (int ms = 0; ms < 4; ++ms) {
                acc[ms] = __builtin_amdgcn_wmma_f32_16x16x32_bf16(false, ah[ms], false, bh, (short)0, acc[ms], false, false);
                acc[ms] = __builtin_amdgcn_wmma_f32_16x16x32_bf16(false, ah[ms], false, bl, (short)0, acc[ms], false, false);
                acc[ms] = __builtin_amdgcn_wmma_f32_16x16x32_bf16(false, al[ms], false, bh, (short)0, acc[ms], false, false);
            }
        }
    }

    // C/D layout: lane<16 -> M = i, lane>=16 -> M = i+8
#pragma unroll
    for (int ms = 0; ms < 4; ++ms) {
#pragma unroll
        for (int i = 0; i < 8; ++i) {
            const int lm = ms * 16 + i + laneHalf * 8;
            const int ti = tileBase + lm;
            if (ti < cnt) {
                unsafeAtomicAdd(&combined[(size_t)s_tok[lm] * DM + ncol],
                                s_gate[lm] * __expf(acc[ms][i]));
            }
        }
    }
}

// ---------------- kernel: final log combine ----------------
__global__ void combine_kernel(const float* __restrict__ comb, float* __restrict__ out, int n) {
    int i = blockIdx.x * blockDim.x + threadIdx.x;
    if (i >= n) return;
    float c = comb[i];
    const float eps = 2.2204460492503131e-16f;   // float32(np.finfo(float64).eps)
    out[i] = __logf(c == 0.0f ? eps : c);
}

// ---------------- host launcher ----------------
extern "C" void kernel_launch(void* const* d_in, const int* in_sizes, int n_in,
                              void* d_out, int out_size, void* d_ws, size_t ws_size,
                              hipStream_t stream) {
    const float* x  = (const float*)d_in[0];   // [N_TOK, DM]
    const float* wg = (const float*)d_in[1];   // [DM, NE]
    const float* ew = (const float*)d_in[2];   // [NE, DM, DM]
    (void)in_sizes; (void)n_in; (void)out_size; (void)ws_size;

    char* ws = (char*)d_ws;
    float*  combined = (float*)(ws + OFF_COMBINED);
    __bf16* xhi      = (__bf16*)(ws + OFF_XHI);
    __bf16* xlo      = (__bf16*)(ws + OFF_XLO);
    __bf16* whT      = (__bf16*)(ws + OFF_WHT);
    __bf16* wlT      = (__bf16*)(ws + OFF_WLT);
    int*    counts   = (int*)(ws + OFF_CNT);
    float*  imp      = (float*)(ws + OFF_IMP);
    float*  ldc      = (float*)(ws + OFF_LDC);
    int*    btok     = (int*)(ws + OFF_BTOK);
    float*  bgate    = (float*)(ws + OFF_BGATE);

    float* out      = (float*)d_out;                 // [N_TOK*DM] + 1 (loss)
    float* out_loss = out + (size_t)N_TOK * DM;

    // zero the accumulator + counters (graph-capture safe)
    hipMemsetAsync(combined, 0, (size_t)N_TOK * DM * sizeof(float), stream);
    hipMemsetAsync(ws + OFF_CNT, 0, 1024, stream);

    // precision split / transpose conversions
    {
        int n = N_TOK * DM;
        cvt_x_kernel<<<(n + 255) / 256, 256, 0, stream>>>(x, xhi, xlo, n);
        int m = NE * DM * DM;
        cvt_w_kernel<<<(m + 255) / 256, 256, 0, stream>>>(ew, whT, wlT);
    }

    // gating + bucketing
    gate_kernel<<<N_TOK / 256, 256, 0, stream>>>(x, wg, counts, imp, ldc, btok, bgate);

    // balance loss scalar
    loss_kernel<<<1, 32, 0, stream>>>(imp, ldc, out_loss);

    // bucketed expert GEMMs (worst-case grid; blocks past the per-expert count exit immediately)
    {
        dim3 grid(N_TOK / MT, DM / 128, NE);
        moe_gemm_kernel<<<grid, 256, 0, stream>>>(xhi, xlo, whT, wlT,
                                                  counts, btok, bgate, combined);
    }

    // out = log(combined), eps where zero
    {
        int n = N_TOK * DM;
        combine_kernel<<<(n + 255) / 256, 256, 0, stream>>>(combined, out, n);
    }
}
